// NaiveGNN_69363721831097
// MI455X (gfx1250) — matrix operand order repaired and checked by reading, not compiled, not run on hardware
//
#include <hip/hip_runtime.h>
#include <math.h>

typedef __attribute__((ext_vector_type(16))) _Float16 v16h;
typedef __attribute__((ext_vector_type(8)))  float    v8f;

#define N_CELLS 300000
#define N_NETS  100000
#define N_PINS  1200000
#define N_PT    1000000
#define HC 64
#define HN 64
#define HP 16
#define RN 4
#define RP 6

static __device__ __forceinline__ float ssp_f(float x) {
    // ShiftedSoftplus: softplus(x) - log(2), numerically stable
    float sp = fmaxf(x, 0.0f) + log1pf(__expf(-fabsf(x)));
    return sp - 0.69314718055994530942f;
}

static __device__ __forceinline__ v8f wmma_f16(v16h a, v16h b, v8f c) {
    // D(16x16,f32) = A(16x32,f16) * B(32x16,f16) + C
    return __builtin_amdgcn_wmma_f32_16x16x32_f16(
        /*neg_a=*/false, a, /*neg_b=*/false, b,
        /*c_mod=*/(short)0, c, /*reuse_a=*/false, /*reuse_b=*/false);
}

// ---------------------------------------------------------------------------
// Pack f32 weight matrices [K x 64] into f16 WMMA B-matrix fragments.
// Fragment layout (per ISA 16-bit B 32x16): lane L holds column n = t*16+(L&15);
// element e (0..15) holds K = kb*32 + (L>>4)*8 + e (e<8), kb*32 + 16 + (L>>4)*8 + (e-8) (e>=8).  Fragment f = t*nkb + kb,
// stored contiguously: pk[(f*32 + L)*16 + e].
// ---------------------------------------------------------------------------
__global__ void k_pack(const float* __restrict__ We1, const float* __restrict__ We2,
                       const float* __restrict__ Wo, const float* __restrict__ Wn,
                       _Float16* __restrict__ we1pk, _Float16* __restrict__ we2pk,
                       _Float16* __restrict__ wopk, _Float16* __restrict__ wnpk) {
    int idx = blockIdx.x * blockDim.x + threadIdx.x;
    if (idx < 4 * 32 * 16) {          // We1: K_actual=16 (padded to 32), nkb=1
        int e = idx & 15, L = (idx >> 4) & 31, t = idx >> 9;
        int n = t * 16 + (L & 15);
        int K = (e < 8) ? ((L >> 4) * 8 + e) : (16 + (L >> 4) * 8 + (e - 8));  // kb = 0
        we1pk[idx] = (K < HP) ? (_Float16)We1[K * HC + n] : (_Float16)0.0f;
    }
    if (idx < 8 * 32 * 16) {          // We2 / Wo / Wn: K_actual=64, nkb=2
        int e = idx & 15, L = (idx >> 4) & 31, f = idx >> 9;
        int t = f >> 1, kb = f & 1;
        int n = t * 16 + (L & 15);
        int K = kb * 32 + ((e < 8) ? ((L >> 4) * 8 + e) : (16 + (L >> 4) * 8 + (e - 8)));
        we2pk[idx] = (_Float16)We2[K * HC + n];
        wopk[idx]  = (_Float16)Wo[K * HC + n];
        wnpk[idx]  = (_Float16)Wn[K * HC + n];
    }
}

__global__ void k_zero(float* __restrict__ p, long n) {
    long i = (long)blockIdx.x * blockDim.x + threadIdx.x;
    long stride = (long)gridDim.x * blockDim.x;
    for (; i < n; i += stride) p[i] = 0.0f;
}

// ---------------------------------------------------------------------------
// node_proj = tanh(net_feat @ W_net + b_net) @ Wn      [N_NETS x 64]
// One wave per 16 nets; each lane computes its 32 A-layout activations
// (each (row,k) produced exactly once), then 8 WMMAs against packed Wn.
// ---------------------------------------------------------------------------
__global__ void __launch_bounds__(256)
k_netproj(const float* __restrict__ net_feat, const float* __restrict__ W_net,
          const float* __restrict__ b_net, const _Float16* __restrict__ wnpk,
          float* __restrict__ node_proj) {
    const int tid = threadIdx.x;
    const int lane = tid & 31;
    const int m = lane & 15, g = lane >> 4;
    const long tile = (long)blockIdx.x * 8 + (tid >> 5);
    const long base = tile * 16;
    if (base >= N_NETS) return;

    const float* nf = net_feat + (base + m) * RN;
    float f0 = nf[0], f1 = nf[1], f2 = nf[2], f3 = nf[3];
    v16h a[2];
    #pragma unroll
    for (int kb = 0; kb < 2; ++kb) {
        #pragma unroll
        for (int j = 0; j < 8; ++j) {
            int k0 = kb * 32 + g * 8 + j;
            int k1 = k0 + 16;
            float h0 = tanhf(b_net[k0] + f0 * W_net[0 * HN + k0] + f1 * W_net[1 * HN + k0]
                                       + f2 * W_net[2 * HN + k0] + f3 * W_net[3 * HN + k0]);
            float h1 = tanhf(b_net[k1] + f0 * W_net[0 * HN + k1] + f1 * W_net[1 * HN + k1]
                                       + f2 * W_net[2 * HN + k1] + f3 * W_net[3 * HN + k1]);
            a[kb][j]     = (_Float16)h0;
            a[kb][j + 8] = (_Float16)h1;
        }
    }
    #pragma unroll
    for (int t = 0; t < 4; ++t) {
        v8f c = {};
        #pragma unroll
        for (int kb = 0; kb < 2; ++kb) {
            v16h b = *(const v16h*)(wnpk + (((t * 2 + kb) * 32 + lane) << 4));
            c = wmma_f16(a[kb], b, c);
        }
        int n = t * 16 + m;   // Wn has no bias (project_node)
        #pragma unroll
        for (int v = 0; v < 8; ++v)
            node_proj[(base + v + 8 * g) * HC + n] = c[v];
    }
}

// ---------------------------------------------------------------------------
// Fused per-pin pipeline, one wave per 16-pin tile, 12 WMMAs per tile:
//   h   = tanh(pin_feat @ W_pin + b_pin)            (6 -> 16, VALU)
//   E1  = ssp(h @ We1 + be1)                        (16 -> 64, 4 WMMA)
//   E2  = ssp(E1 @ We2 + be2)                       (64 -> 64, 8 WMMA)
//   agg_cell[src] += node_proj[dst] * E2            (gather + atomic scatter)
// ---------------------------------------------------------------------------
__global__ void __launch_bounds__(256)
k_edges(const float* __restrict__ pin_feat, const int* __restrict__ pins_src,
        const int* __restrict__ pins_dst, const float* __restrict__ W_pin,
        const float* __restrict__ b_pin, const float* __restrict__ be1,
        const float* __restrict__ be2, const _Float16* __restrict__ we1pk,
        const _Float16* __restrict__ we2pk, const float* __restrict__ node_proj,
        float* __restrict__ agg_cell) {
    __shared__ _Float16 sE1[8][16 * HC];   // per-wave 16x64 f16 tile
    const int tid = threadIdx.x;
    const int w = tid >> 5, lane = tid & 31;
    const int m = lane & 15, g = lane >> 4;
    const long tile = (long)blockIdx.x * 8 + w;   // grid sized exactly: no tail
    const long base = tile * 16;

    // --- input projection: this lane covers row m, features k = g*8..g*8+7 ---
    const float* pf = pin_feat + (base + m) * RP;
    float p0 = pf[0], p1 = pf[1], p2 = pf[2], p3 = pf[3], p4 = pf[4], p5 = pf[5];
    v16h a1;
    #pragma unroll
    for (int j = 0; j < 8; ++j) {
        int k = g * 8 + j;
        float a = b_pin[k] + p0 * W_pin[0 * HP + k] + p1 * W_pin[1 * HP + k]
                           + p2 * W_pin[2 * HP + k] + p3 * W_pin[3 * HP + k]
                           + p4 * W_pin[4 * HP + k] + p5 * W_pin[5 * HP + k];
        a1[j] = (_Float16)tanhf(a);
        a1[j + 8] = (_Float16)0.0f;   // K = 16..31 zero padding
    }

    // --- layer 1: E1(16x64) = A1(16x32) x We1 ---
    v8f acc1[4];
    #pragma unroll
    for (int t = 0; t < 4; ++t) {
        v16h b = *(const v16h*)(we1pk + ((t * 32 + lane) << 4));
        v8f c = {};
        acc1[t] = wmma_f16(a1, b, c);
    }
    // bias + ssp, park in LDS (row-major f16) to re-fragment as A for layer 2
    _Float16* lds = &sE1[w][0];
    #pragma unroll
    for (int t = 0; t < 4; ++t) {
        int n = t * 16 + m;          // D layout: N = lane&15 (+16t)
        float bias = be1[n];
        #pragma unroll
        for (int v = 0; v < 8; ++v) {
            int M = v + 8 * g;       // D layout: M = vgpr + 8*(lane>>4)
            lds[M * HC + n] = (_Float16)ssp_f(acc1[t][v] + bias);
        }
    }
    __syncthreads();

    // --- layer 2: build A fragments from LDS (ISA 16-bit A layout) ---
    v16h a2[2];
    #pragma unroll
    for (int kb = 0; kb < 2; ++kb) {
        #pragma unroll
        for (int j = 0; j < 8; ++j) {
            a2[kb][j]     = lds[m * HC + kb * 32 + g * 8 + j];
            a2[kb][j + 8] = lds[m * HC + kb * 32 + 16 + g * 8 + j];
        }
    }
    v8f acc2[4];
    #pragma unroll
    for (int t = 0; t < 4; ++t) {
        v8f c = {};
        #pragma unroll
        for (int kb = 0; kb < 2; ++kb) {
            v16h b = *(const v16h*)(we2pk + (((t * 2 + kb) * 32 + lane) << 4));
            c = wmma_f16(a2[kb], b, c);
        }
        acc2[t] = c;
    }

    // --- ssp, gather node_proj[dst], atomic scatter into agg_cell[src] ---
    int dsts[8], srcs[8];
    #pragma unroll
    for (int v = 0; v < 8; ++v) {
        long e = base + v + 8 * g;
        dsts[v] = pins_dst[e];
        srcs[v] = pins_src[e];
    }
    #pragma unroll
    for (int t = 0; t < 4; ++t) {
        int n = t * 16 + m;
        float bias = be2[n];
        #pragma unroll
        for (int v = 0; v < 8; ++v) {
            float ep = ssp_f(acc2[t][v] + bias);
            float msg = ep * node_proj[(long)dsts[v] * HC + n];
            atomicAdd(&agg_cell[(long)srcs[v] * HC + n], msg);
        }
    }
}

// ---------------------------------------------------------------------------
// new_cell = ssp(agg_cell @ Wo + bo), one wave per 16 cells, 8 WMMAs/tile.
// ---------------------------------------------------------------------------
__global__ void __launch_bounds__(256)
k_cells(const float* __restrict__ agg_cell, const _Float16* __restrict__ wopk,
        const float* __restrict__ bo, float* __restrict__ new_cell) {
    const int tid = threadIdx.x;
    const int lane = tid & 31;
    const int m = lane & 15, g = lane >> 4;
    const long tile = (long)blockIdx.x * 8 + (tid >> 5);
    const long base = tile * 16;
    if (base >= N_CELLS) return;

    const float* arow = agg_cell + (base + m) * HC;
    v16h a[2];
    #pragma unroll
    for (int kb = 0; kb < 2; ++kb) {
        #pragma unroll
        for (int j = 0; j < 8; ++j) {
            a[kb][j]     = (_Float16)arow[kb * 32 + g * 8 + j];
            a[kb][j + 8] = (_Float16)arow[kb * 32 + 16 + g * 8 + j];
        }
    }
    #pragma unroll
    for (int t = 0; t < 4; ++t) {
        v8f c = {};
        #pragma unroll
        for (int kb = 0; kb < 2; ++kb) {
            v16h b = *(const v16h*)(wopk + (((t * 2 + kb) * 32 + lane) << 4));
            c = wmma_f16(a[kb], b, c);
        }
        int n = t * 16 + m;
        float bias = bo[n];
        #pragma unroll
        for (int v = 0; v < 8; ++v)
            new_cell[(base + v + 8 * g) * HC + n] = ssp_f(c[v] + bias);
    }
}

// ---------------------------------------------------------------------------
// Edge readout over points-to pairs (L2-resident gathers of new_cell rows).
// Kept in f32 VALU: gather-bound, and exp(12*tanh(.)) is too steep for f16.
// ---------------------------------------------------------------------------
__global__ void __launch_bounds__(256)
k_readout(const int* __restrict__ pt_f, const int* __restrict__ pt_s,
          const float* __restrict__ new_cell, const float* __restrict__ W_dis,
          const float* __restrict__ b_dis, const float* __restrict__ W_ang,
          const float* __restrict__ b_ang, float* __restrict__ out) {
    __shared__ float sd[2 * HC], sa[2 * HC];
    for (int i = threadIdx.x; i < 2 * HC; i += blockDim.x) { sd[i] = W_dis[i]; sa[i] = W_ang[i]; }
    __syncthreads();
    long e = (long)blockIdx.x * blockDim.x + threadIdx.x;
    if (e >= N_PT) return;
    const float* rf = new_cell + (long)pt_f[e] * HC;
    const float* rs = new_cell + (long)pt_s[e] * HC;
    float d = b_dis[0], a = b_ang[0];
    #pragma unroll 8
    for (int k = 0; k < HC; ++k) { float v = rf[k]; d += v * sd[k]; a += v * sa[k]; }
    #pragma unroll 8
    for (int k = 0; k < HC; ++k) { float v = rs[k]; d += v * sd[HC + k]; a += v * sa[HC + k]; }
    out[e] = __expf(12.0f * tanhf(d));
    out[N_PT + e] = tanhf(a) * 4.0f;
}

extern "C" void kernel_launch(void* const* d_in, const int* in_sizes, int n_in,
                              void* d_out, int out_size, void* d_ws, size_t ws_size,
                              hipStream_t stream) {
    (void)in_sizes; (void)n_in; (void)out_size; (void)ws_size;
    // Input order per setup_inputs(); GraphConv branch (new_net) is dead code,
    // so cell_feat/W_cell/b_cell/W_gc/b_gc are intentionally unused.
    const float* net_feat = (const float*)d_in[1];
    const float* pin_feat = (const float*)d_in[2];
    const int*   pins_src = (const int*)d_in[3];
    const int*   pins_dst = (const int*)d_in[4];
    const int*   pt_f     = (const int*)d_in[5];
    const int*   pt_s     = (const int*)d_in[6];
    const float* W_net    = (const float*)d_in[9];
    const float* b_net    = (const float*)d_in[10];
    const float* W_pin    = (const float*)d_in[11];
    const float* b_pin    = (const float*)d_in[12];
    const float* We1      = (const float*)d_in[15];
    const float* be1      = (const float*)d_in[16];
    const float* We2      = (const float*)d_in[17];
    const float* be2      = (const float*)d_in[18];
    const float* Wn       = (const float*)d_in[19];
    const float* Wo       = (const float*)d_in[20];
    const float* bo       = (const float*)d_in[21];
    const float* W_dis    = (const float*)d_in[22];
    const float* b_dis    = (const float*)d_in[23];
    const float* W_ang    = (const float*)d_in[24];
    const float* b_ang    = (const float*)d_in[25];

    // Workspace layout (bytes): packed f16 weights + f32 intermediates (~179 MB)
    char* ws = (char*)d_ws;
    _Float16* we1pk = (_Float16*)(ws + 0);        // 4096 B
    _Float16* we2pk = (_Float16*)(ws + 4096);     // 8192 B
    _Float16* wopk  = (_Float16*)(ws + 12288);    // 8192 B
    _Float16* wnpk  = (_Float16*)(ws + 20480);    // 8192 B
    float* node_proj = (float*)(ws + 28672);
    float* agg_cell  = (float*)(ws + 28672 + (size_t)N_NETS  * HC * 4);
    float* new_cell  = (float*)(ws + 28672 + (size_t)N_NETS  * HC * 4
                                            + (size_t)N_CELLS * HC * 4);

    k_pack<<<16, 256, 0, stream>>>(We1, We2, Wo, Wn, we1pk, we2pk, wopk, wnpk);
    k_zero<<<2048, 256, 0, stream>>>(agg_cell, (long)N_CELLS * HC);
    // 100k nets / 16 per wave / 8 waves per block = 781.25 -> 782 blocks (tail guard)
    k_netproj<<<(N_NETS / 16 + 7) / 8, 256, 0, stream>>>(net_feat, W_net, b_net,
                                                         wnpk, node_proj);
    // 1.2M pins / 16 per wave / 8 waves per block = 9375 blocks (exact)
    k_edges<<<N_PINS / 16 / 8, 256, 0, stream>>>(pin_feat, pins_src, pins_dst,
                                                 W_pin, b_pin, be1, be2,
                                                 we1pk, we2pk, node_proj, agg_cell);
    k_cells<<<(N_CELLS / 16 + 7) / 8, 256, 0, stream>>>(agg_cell, wopk, bo, new_cell);
    k_readout<<<(N_PT + 255) / 256, 256, 0, stream>>>(pt_f, pt_s, new_cell,
                                                      W_dis, b_dis, W_ang, b_ang,
                                                      (float*)d_out);
}
